// BDH_33569464385660
// MI455X (gfx1250) — matrix operand order repaired and checked by reading, not compiled, not run on hardware
//
#include <hip/hip_runtime.h>
#include <hip/hip_bf16.h>

// ---------------------------------------------------------------------------
// Types for CDNA5 WMMA (wave32): v_wmma_f32_16x16x32_bf16
// ---------------------------------------------------------------------------
typedef __bf16 bhalf;
typedef __attribute__((ext_vector_type(16))) __bf16 v16bf;
typedef __attribute__((ext_vector_type(8)))  __bf16 v8bf;
typedef __attribute__((ext_vector_type(8)))  float  v8f;

union FragCvt { struct { v8bf lo, hi; } h; v16bf v; };

// A fragment: 16x32 bf16, row-major source with leading dim ld (elements).
// lane<16 : row=lane,    K = {0..7} U {16..23}
// lane>=16: row=lane-16, K = {8..15} U {24..31}
__device__ __forceinline__ v16bf load_a_frag(const bhalf* base, int ld, int lane) {
    int r  = lane & 15;
    int ko = (lane >> 4) << 3;           // 0 or 8
    const bhalf* p = base + r * ld + ko;
    FragCvt f;
    f.h.lo = *(const v8bf*)(p);          // K: ko .. ko+7
    f.h.hi = *(const v8bf*)(p + 16);     // K: ko+16 .. ko+23
    return f.v;
}

// B fragment: 32x16 bf16 where the source is B^T stored [N][K] row-major.
// lane<16 : col=lane,    K = {0..7} U {16..23}
// lane>=16: col=lane-16, K = {8..15} U {24..31}
__device__ __forceinline__ v16bf load_b_frag(const bhalf* bt, int ld, int lane) {
    int c  = lane & 15;
    int ko = (lane >> 4) << 3;           // 0 or 8
    const bhalf* p = bt + c * ld + ko;
    FragCvt f;
    f.h.lo = *(const v8bf*)(p);
    f.h.hi = *(const v8bf*)(p + 16);
    return f.v;
}

__device__ __forceinline__ v8f wmma_bf16(v16bf a, v16bf b, v8f c) {
    return __builtin_amdgcn_wmma_f32_16x16x32_bf16(false, a, false, b,
                                                   (short)0, c, false, false);
}

// ---------------------------------------------------------------------------
// Model constants
// ---------------------------------------------------------------------------
#define BB   2
#define TT   2048
#define DD   512
#define NHH  8
#define NN   256          // per-head latent
#define CC   2048         // NH*NN
#define BT   (BB*TT)      // 4096

// ---------------------------------------------------------------------------
// Weight transpose + f32->bf16:  dst[b][c][r] = src[b][r][c]
// ---------------------------------------------------------------------------
__global__ void k_tconv(const float* __restrict__ src, bhalf* __restrict__ dst,
                        int R, int C) {
    long base = (long)blockIdx.y * R * C;
    int i = blockIdx.x * 256 + threadIdx.x;
    if (i >= R * C) return;
    int r = i % R, c = i / R;
    dst[base + i] = (bhalf)src[base + (long)r * C + c];
}

__global__ void k_cvt(const float* __restrict__ src, bhalf* __restrict__ dst, int n) {
    int i = blockIdx.x * 256 + threadIdx.x;
    if (i < n) dst[i] = (bhalf)src[i];
}

// ---------------------------------------------------------------------------
// Generic GEMM: C[M][N] = A[M][K] * B^T[N][K]  (+bias, +relu)
// block = 256 threads = 8 waves; block tile 128x128; wave tile 64x32
// flags: bit0 = add bias, bit1 = relu
// ---------------------------------------------------------------------------
__global__ __launch_bounds__(256)
void k_gemm(const bhalf* __restrict__ A, const bhalf* __restrict__ BT_,
            float* __restrict__ C, const float* __restrict__ bias,
            int M, int N, int K, int flags) {
    int wave = threadIdx.x >> 5, lane = threadIdx.x & 31;
    int wm = blockIdx.x * 128 + (wave & 1) * 64;
    int wn = blockIdx.y * 128 + (wave >> 1) * 32;

    v8f acc[4][2] = {};
    for (int k = 0; k < K; k += 32) {
        v16bf bf0 = load_b_frag(BT_ + (long)(wn +  0) * K + k, K, lane);
        v16bf bf1 = load_b_frag(BT_ + (long)(wn + 16) * K + k, K, lane);
#pragma unroll
        for (int i = 0; i < 4; ++i) {
            v16bf af = load_a_frag(A + (long)(wm + i * 16) * K + k, K, lane);
            acc[i][0] = wmma_bf16(af, bf0, acc[i][0]);
            acc[i][1] = wmma_bf16(af, bf1, acc[i][1]);
        }
    }
    int rl = (lane >> 4) * 8, cl = lane & 15;
#pragma unroll
    for (int i = 0; i < 4; ++i)
#pragma unroll
        for (int j = 0; j < 2; ++j) {
            int row0 = wm + i * 16 + rl;
            int col  = wn + j * 16 + cl;
            float bv = (flags & 1) ? bias[col] : 0.0f;
#pragma unroll
            for (int e = 0; e < 8; ++e) {
                float v = acc[i][j][e] + bv;
                if (flags & 2) v = fmaxf(v, 0.0f);
                C[(long)(row0 + e) * N + col] = v;
            }
        }
}

// ---------------------------------------------------------------------------
// RoPE: x_sparse f32 [B][T][C] -> QR bf16 [B][NH][T][NN]
// freqs: q = 2*floor(n/2); f = 2^(-16*q/256) / (2*pi); angle = frac(t*f)*2pi
// ---------------------------------------------------------------------------
__global__ void k_rope(const float* __restrict__ xsp, bhalf* __restrict__ qr) {
    int idx = blockIdx.x * 256 + threadIdx.x;        // pair index
    if (idx >= BB * TT * NHH * (NN / 2)) return;
    int np = idx & 127;
    int h  = (idx >> 7) & 7;
    int t  = (idx >> 10) & (TT - 1);
    int b  = idx >> 21;
    int n0 = np * 2;
    float freq = exp2f(-16.0f * (float)n0 * (1.0f / 256.0f)) * 0.15915494309189535f;
    float ph = (float)t * freq;
    ph -= floorf(ph);
    float ang = ph * 6.283185307179586f;
    float s, c;
    __sincosf(ang, &s, &c);
    const float* xp = xsp + ((long)(b * TT + t) * CC) + h * NN + n0;
    float v0 = xp[0], v1 = xp[1];
    bhalf* qp = qr + (((long)(b * NHH + h) * TT + t) * NN) + n0;
    qp[0] = (bhalf)(v0 * c - v1 * s);
    qp[1] = (bhalf)(v1 * c + v0 * s);
}

// ---------------------------------------------------------------------------
// Fused causal "attention": Y = (strict_lower ⊙ (QR·QRᵀ)) · Xs   (no softmax)
//   QR : bf16 [B][NH][T][NN]  (row-major per (b,h))  -> A and Bᵀ of phase A
//   xsT: bf16 [B][DD][T]      (transposed xs)        -> Bᵀ of phase B
//   Y  : f32  [B][NH][T][DD]
// WG = 256 thr = 8 waves handles 64 rows x 512 cols; S tile 64x64 in LDS.
// ---------------------------------------------------------------------------
__global__ __launch_bounds__(256)
void k_attn(const bhalf* __restrict__ QR, const bhalf* __restrict__ xsT,
            float* __restrict__ yKV) {
    int bh = blockIdx.y;
    int b  = bh >> 3;
    int r0 = blockIdx.x * 64;
    int wave = threadIdx.x >> 5, lane = threadIdx.x & 31;
    const bhalf* q  = QR  + (long)bh * TT * NN;
    const bhalf* xt = xsT + (long)b  * DD * TT;

    __shared__ __align__(16) bhalf S[64 * 64];

    int wmi = wave >> 1;          // phase-B row subtile (0..3)
    int wni = wave & 1;           // phase-B column half (0..1)
    int rl = (lane >> 4) * 8, cl = lane & 15;

    v8f Y[16] = {};

    for (int s0 = 0; s0 <= r0; s0 += 64) {
        // ---- phase A: S = mask(Q_r · Q_sᵀ), each wave 2 of 16 subtiles ----
#pragma unroll
        for (int t2 = 0; t2 < 2; ++t2) {
            int st = wave * 2 + t2;
            int sm = st >> 2, sn = st & 3;
            v8f sacc = {};
            const bhalf* Ap = q + (long)(r0 + sm * 16) * NN;
            const bhalf* Bp = q + (long)(s0 + sn * 16) * NN;
#pragma unroll
            for (int k = 0; k < NN; k += 32) {
                v16bf af  = load_a_frag(Ap + k, NN, lane);
                v16bf bfr = load_b_frag(Bp + k, NN, lane);
                sacc = wmma_bf16(af, bfr, sacc);
            }
#pragma unroll
            for (int e = 0; e < 8; ++e) {
                int ml = sm * 16 + e + rl;
                int nl = sn * 16 + cl;
                float v = ((r0 + ml) > (s0 + nl)) ? sacc[e] : 0.0f;  // strict lower
                S[ml * 64 + nl] = (bhalf)v;
            }
        }
        __syncthreads();
        // ---- phase B: Y += S · Xs_s  (Bᵀ = xsT, contiguous along T) ----
#pragma unroll
        for (int ks = 0; ks < 64; ks += 32) {
            v16bf af = load_a_frag(S + wmi * 16 * 64 + ks, 64, lane);
#pragma unroll
            for (int j = 0; j < 16; ++j) {
                int d0 = wni * 256 + j * 16;
                v16bf bfr = load_b_frag(xt + (long)d0 * TT + s0 + ks, TT, lane);
                Y[j] = wmma_bf16(af, bfr, Y[j]);
            }
        }
        __syncthreads();
    }

    float* yo = yKV + (long)bh * TT * DD;
#pragma unroll
    for (int j = 0; j < 16; ++j) {
        int col  = wni * 256 + j * 16 + cl;
        int row0 = r0 + wmi * 16 + rl;
#pragma unroll
        for (int e = 0; e < 8; ++e)
            yo[(long)(row0 + e) * DD + col] = Y[j][e];
    }
}

// ---------------------------------------------------------------------------
// LayerNorm over rows of length 512, one wave per row.
//  res==null : out = ln(in)
//  res!=null : out = ln(res + ln(in))     (residual + double LN)
// Optional outputs: f32, bf16, and transposed bf16 [b][d][T].
// ---------------------------------------------------------------------------
__global__ __launch_bounds__(256)
void k_ln(const float* __restrict__ in, const float* __restrict__ res,
          float* __restrict__ out_f, bhalf* __restrict__ out_bf,
          bhalf* __restrict__ out_bfT, int rows) {
    int row  = blockIdx.x * 8 + (threadIdx.x >> 5);
    int lane = threadIdx.x & 31;
    if (row >= rows) return;
    const float* p = in + (long)row * DD;

    float v[16];
#pragma unroll
    for (int i = 0; i < 4; ++i) {
        float4 f = ((const float4*)p)[i * 32 + lane];
        v[i * 4 + 0] = f.x; v[i * 4 + 1] = f.y; v[i * 4 + 2] = f.z; v[i * 4 + 3] = f.w;
    }
    float s1 = 0.f, s2 = 0.f;
#pragma unroll
    for (int i = 0; i < 16; ++i) { s1 += v[i]; s2 += v[i] * v[i]; }
#pragma unroll
    for (int m = 16; m >= 1; m >>= 1) { s1 += __shfl_xor(s1, m, 32); s2 += __shfl_xor(s2, m, 32); }
    float mu = s1 * (1.0f / 512.0f);
    float rs = rsqrtf(s2 * (1.0f / 512.0f) - mu * mu + 1e-5f);
#pragma unroll
    for (int i = 0; i < 16; ++i) v[i] = (v[i] - mu) * rs;

    if (res) {
        const float* rp = res + (long)row * DD;
#pragma unroll
        for (int i = 0; i < 4; ++i) {
            float4 f = ((const float4*)rp)[i * 32 + lane];
            v[i * 4 + 0] += f.x; v[i * 4 + 1] += f.y; v[i * 4 + 2] += f.z; v[i * 4 + 3] += f.w;
        }
        s1 = 0.f; s2 = 0.f;
#pragma unroll
        for (int i = 0; i < 16; ++i) { s1 += v[i]; s2 += v[i] * v[i]; }
#pragma unroll
        for (int m = 16; m >= 1; m >>= 1) { s1 += __shfl_xor(s1, m, 32); s2 += __shfl_xor(s2, m, 32); }
        mu = s1 * (1.0f / 512.0f);
        rs = rsqrtf(s2 * (1.0f / 512.0f) - mu * mu + 1e-5f);
#pragma unroll
        for (int i = 0; i < 16; ++i) v[i] = (v[i] - mu) * rs;
    }

    if (out_f) {
        float* q = out_f + (long)row * DD;
#pragma unroll
        for (int i = 0; i < 4; ++i) {
            float4 f; f.x = v[i*4+0]; f.y = v[i*4+1]; f.z = v[i*4+2]; f.w = v[i*4+3];
            ((float4*)q)[i * 32 + lane] = f;
        }
    }
    if (out_bf) {
        bhalf* q = out_bf + (long)row * DD;
#pragma unroll
        for (int i = 0; i < 16; ++i) {
            int d = ((i >> 2) * 32 + lane) * 4 + (i & 3);
            q[d] = (bhalf)v[i];
        }
    }
    if (out_bfT) {            // [b][d][T]
        int b = row >> 11, t = row & (TT - 1);
#pragma unroll
        for (int i = 0; i < 16; ++i) {
            int d = ((i >> 2) * 32 + lane) * 4 + (i & 3);
            out_bfT[((long)(b * DD + d)) * TT + t] = (bhalf)v[i];
        }
    }
}

// ---------------------------------------------------------------------------
// Second projection + gate: xy[b][t][h*NN+n] =
//     bf16( relu( yln[b,h] · encVᵀ[h] )[t][n] * x_sparse[b][t][h*NN+n] )
// per (b,h): M=T, N=NN=256, K=512
// ---------------------------------------------------------------------------
__global__ __launch_bounds__(256)
void k_yenc_xy(const bhalf* __restrict__ yln,    // [B*NH*T][512]
               const bhalf* __restrict__ encVT,  // [NH*NN][512]
               const float* __restrict__ xsp,    // [B*T][C]
               bhalf* __restrict__ xy) {         // [B*T][C]
    int bh = blockIdx.z;
    int b = bh >> 3, h = bh & 7;
    int wave = threadIdx.x >> 5, lane = threadIdx.x & 31;
    int tm = blockIdx.x * 128 + (wave & 1) * 64;
    int nn = blockIdx.y * 128 + (wave >> 1) * 32;
    const bhalf* A   = yln   + (long)bh * TT * DD;
    const bhalf* BT_ = encVT + (long)(h * NN) * DD;

    v8f acc[4][2] = {};
    for (int k = 0; k < DD; k += 32) {
        v16bf bf0 = load_b_frag(BT_ + (long)(nn +  0) * DD + k, DD, lane);
        v16bf bf1 = load_b_frag(BT_ + (long)(nn + 16) * DD + k, DD, lane);
#pragma unroll
        for (int i = 0; i < 4; ++i) {
            v16bf af = load_a_frag(A + (long)(tm + i * 16) * DD + k, DD, lane);
            acc[i][0] = wmma_bf16(af, bf0, acc[i][0]);
            acc[i][1] = wmma_bf16(af, bf1, acc[i][1]);
        }
    }
    int rl = (lane >> 4) * 8, cl = lane & 15;
#pragma unroll
    for (int i = 0; i < 4; ++i)
#pragma unroll
        for (int j = 0; j < 2; ++j) {
            int c = h * NN + nn + j * 16 + cl;
#pragma unroll
            for (int e = 0; e < 8; ++e) {
                int t = tm + i * 16 + rl + e;
                long ro = (long)(b * TT + t) * CC + c;
                float v = fmaxf(acc[i][j][e], 0.0f) * xsp[ro];
                xy[ro] = (bhalf)v;
            }
        }
}

// ---------------------------------------------------------------------------
// Host launcher
// ---------------------------------------------------------------------------
extern "C" void kernel_launch(void* const* d_in, const int* in_sizes, int n_in,
                              void* d_out, int out_size, void* d_ws, size_t ws_size,
                              hipStream_t stream) {
    const float* x     = (const float*)d_in[0];
    const float* w_in  = (const float*)d_in[1];
    const float* b_in  = (const float*)d_in[2];
    const float* enc   = (const float*)d_in[3];
    const float* encv  = (const float*)d_in[4];
    const float* decd  = (const float*)d_in[5];
    const float* headw = (const float*)d_in[6];
    const float* headb = (const float*)d_in[7];
    float* out = (float*)d_out;

    char* ws = (char*)d_ws;
    size_t off = 0;
    auto alloc = [&](size_t bytes) -> void* {
        void* p = ws + off;
        off += (bytes + 255) & ~(size_t)255;
        return p;
    };

    bhalf* w_inT = (bhalf*)alloc((size_t)DD * DD * 2);
    bhalf* encT  = (bhalf*)alloc((size_t)CC * DD * 2);
    bhalf* encVT = (bhalf*)alloc((size_t)CC * DD * 2);
    bhalf* decT  = (bhalf*)alloc((size_t)DD * CC * 2);
    bhalf* headT = (bhalf*)alloc((size_t)DD * DD * 2);
    bhalf* x_bf  = (bhalf*)alloc((size_t)BT * DD * 2);
    float* xs    = (float*)alloc((size_t)BT * DD * 4);
    bhalf* xs_bf = (bhalf*)alloc((size_t)BT * DD * 2);
    bhalf* xsT   = (bhalf*)alloc((size_t)BB * DD * TT * 2);
    float* xsp   = (float*)alloc((size_t)BT * CC * 4);
    bhalf* qr    = (bhalf*)alloc((size_t)BB * NHH * TT * NN * 2);
    float* ykv   = (float*)alloc((size_t)BB * NHH * TT * DD * 4);
    bhalf* yln   = (bhalf*)alloc((size_t)BB * NHH * TT * DD * 2);
    bhalf* xy    = (bhalf*)alloc((size_t)BT * CC * 2);
    float* tmpf  = (float*)alloc((size_t)BT * DD * 4);

    // --- weight conversion (transposed bf16 layouts) ---
    k_tconv<<<dim3((DD * DD + 255) / 256, 1), 256, 0, stream>>>(w_in, w_inT, DD, DD);
    k_tconv<<<dim3((DD * NN + 255) / 256, NHH), 256, 0, stream>>>(enc,  encT,  DD, NN);
    k_tconv<<<dim3((DD * NN + 255) / 256, NHH), 256, 0, stream>>>(encv, encVT, DD, NN);
    k_tconv<<<dim3((CC * DD + 255) / 256, 1), 256, 0, stream>>>(decd, decT, CC, DD);
    k_tconv<<<dim3((DD * DD + 255) / 256, 1), 256, 0, stream>>>(headw, headT, DD, DD);
    k_cvt<<<(BT * DD + 255) / 256, 256, 0, stream>>>(x, x_bf, BT * DD);

    // --- input projection + LN ---
    k_gemm<<<dim3(BT / 128, DD / 128), 256, 0, stream>>>(x_bf, w_inT, tmpf, b_in,
                                                         BT, DD, DD, 1);
    k_ln<<<BT / 8, 256, 0, stream>>>(tmpf, nullptr, xs, xs_bf, xsT, BT);

    // --- layers ---
    for (int layer = 0; layer < 3; ++layer) {
        // x_sparse = relu(xs @ encoder)  [BT x C]
        k_gemm<<<dim3(BT / 128, CC / 128), 256, 0, stream>>>(xs_bf, encT, xsp, nullptr,
                                                             BT, CC, DD, 2);
        // QR = rope(x_sparse)
        k_rope<<<(BB * TT * NHH * (NN / 2)) / 256, 256, 0, stream>>>(xsp, qr);
        // yKV = (mask . QR QR^T) @ xs
        k_attn<<<dim3(TT / 64, BB * NHH), 256, 0, stream>>>(qr, xsT, ykv);
        // ln(yKV) -> bf16
        k_ln<<<(BB * NHH * TT) / 8, 256, 0, stream>>>(ykv, nullptr, nullptr, yln,
                                                      nullptr, BB * NHH * TT);
        // xy = bf16(relu(yln @ encoder_v) * x_sparse)
        k_yenc_xy<<<dim3(TT / 128, NN / 128, BB * NHH), 256, 0, stream>>>(yln, encVT,
                                                                          xsp, xy);
        // yMLP = xy @ decoder
        k_gemm<<<dim3(BT / 128, DD / 128), 256, 0, stream>>>(xy, decT, tmpf, nullptr,
                                                             BT, DD, CC, 0);
        // xs = ln(xs + ln(yMLP))
        k_ln<<<BT / 8, 256, 0, stream>>>(tmpf, xs, xs, xs_bf, xsT, BT);
    }

    // --- head ---
    k_gemm<<<dim3(BT / 128, DD / 128), 256, 0, stream>>>(xs_bf, headT, out, headb,
                                                         BT, DD, DD, 1);
}
